// SSMMLayer_22058952032337
// MI455X (gfx1250) — matrix operand order repaired and checked by reading, not compiled, not run on hardware
//
#include <hip/hip_runtime.h>
#include <hip/hip_bf16.h>
#include <math.h>

// ---------------------------------------------------------------------------
// SSM scan for MI455X (gfx1250, wave32, WMMA bf16 16x16x32, f32 accumulate)
//
//   Pass 1 (parallel):  XB[t,b,:] = x[b,t,:] @ Bm^T           (bf16 in ws)
//                       G [t,b,:] = sigmoid(x[b,t,:]@Wg^T+bg) (bf16 in ws)
//   Pass 2 (serial, 512 launches): s = s + mask((s@A^T + XB_t)*dt)*G_t
//   Pass 3 (parallel):  ys = Sbuf @ C^T -> d_out (f32), plus final state
//
// New this round: GLOBAL_LOAD_ASYNC_TO_LDS_B128 staging for bf16 operand
// tiles (ASYNCcnt-tracked, VGPR-free) + double-buffered LDS pipeline.
// ---------------------------------------------------------------------------

typedef __attribute__((ext_vector_type(16))) __bf16 v16bf;
typedef __attribute__((ext_vector_type(8)))  __bf16 bf16x8;
typedef __attribute__((ext_vector_type(8)))  float  v8f;
typedef __attribute__((ext_vector_type(4)))  float  f32x4;

#define HIDDEN 1024
#define BATCH  128
#define SEQ    512

#define BM 128
#define BN 64
#define BK 32
#define LDA 40   // padded LDS row stride (bf16 elems) to spread banks
#define LDB 40

#define EPI_XB   0
#define EPI_GATE 1
#define EPI_STEP 2
#define EPI_OUT  3

// ---- CDNA5 async global->LDS copy (ASYNCcnt path), 16 bytes per lane ----
__device__ __forceinline__ void async_copy_b128(void* lds_dst, const void* gsrc) {
    // low 32 bits of a flat shared address are the LDS byte offset
    unsigned lds = (unsigned)(unsigned long long)lds_dst;
    asm volatile("global_load_async_to_lds_b128 %0, %1, off"
                 :: "v"(lds), "v"(gsrc)
                 : "memory");
}
__device__ __forceinline__ void wait_async0() {
    asm volatile("s_wait_asynccnt 0x0" ::: "memory");
}

__global__ void cvt_f32_to_bf16(const float* __restrict__ src,
                                __bf16* __restrict__ dst, int n) {
    int i = blockIdx.x * blockDim.x + threadIdx.x;
    if (i < n) dst[i] = (__bf16)src[i];
}

// Generic weight-transposed GEMM:  out[m,n] = sum_k Aop[m,k] * W[n,k]
// W is bf16 row-major [N][K]; A operand is f32 (converted) or bf16 (async).
template<int EPI, bool A_IS_BF16>
__global__ __launch_bounds__(256)
void ssm_gemm(const float*  __restrict__ Af32,
              const __bf16* __restrict__ Abf,
              const __bf16* __restrict__ W,
              const __bf16* __restrict__ xb,
              const __bf16* __restrict__ gate,
              const float*  __restrict__ sSrc,
              float*        __restrict__ sDst,
              const float*  __restrict__ bg,
              const float*  __restrict__ log_dt,
              float*        __restrict__ outF,
              __bf16*       __restrict__ outBf,
              float*        __restrict__ stateOut,
              int t)
{
    __shared__ __bf16 As[2][BM * LDA];
    __shared__ __bf16 Bs[2][BN * LDB];

    const int tid   = threadIdx.x;
    const int wave  = tid >> 5;
    const int lane  = tid & 31;
    const int half  = lane >> 4;   // wave32 half
    const int l15   = lane & 15;
    const int waveM = wave >> 1;   // 0..3  -> 32-row strip
    const int waveN = wave & 1;    // 0..1  -> 32-col strip
    const int blockM = blockIdx.y * BM;
    const int blockN = blockIdx.x * BN;

    v8f acc[2][2] = {};

    // ---- cooperative tile loaders ----
    auto loadB = [&](int k0, int buf) {      // 64 x 32 bf16, async to LDS
        const int row = tid >> 2;            // 0..63
        const int col = (tid & 3) * 8;       // 0,8,16,24
        async_copy_b128(&Bs[buf][row * LDB + col],
                        W + (size_t)(blockN + row) * HIDDEN + k0 + col);
    };
    auto loadA = [&](int k0, int buf) {      // 128 x 32 into LDS as bf16
        const int row = tid >> 1;            // 0..127
        const int col = (tid & 1) * 16;      // 0,16
        if (A_IS_BF16) {
            const __bf16* src = Abf + (size_t)(blockM + row) * HIDDEN + k0 + col;
            async_copy_b128(&As[buf][row * LDA + col],     src);
            async_copy_b128(&As[buf][row * LDA + col + 8], src + 8);
        } else {
            const float* src = Af32 + (size_t)(blockM + row) * HIDDEN + k0 + col;
            if (k0 + BK < HIDDEN) __builtin_prefetch(src + BK, 0, 0);
            __bf16 tmp[16];
            #pragma unroll
            for (int i = 0; i < 16; i += 4) {
                f32x4 v = *(const f32x4*)(src + i);
                tmp[i + 0] = (__bf16)v.x; tmp[i + 1] = (__bf16)v.y;
                tmp[i + 2] = (__bf16)v.z; tmp[i + 3] = (__bf16)v.w;
            }
            bf16x8 lo, hi;
            #pragma unroll
            for (int i = 0; i < 8; ++i) { lo[i] = tmp[i]; hi[i] = tmp[8 + i]; }
            *(bf16x8*)&As[buf][row * LDA + col]     = lo;
            *(bf16x8*)&As[buf][row * LDA + col + 8] = hi;
        }
    };

    // ---- prologue: fill buffer 0 ----
    loadB(0, 0);
    loadA(0, 0);
    wait_async0();
    __syncthreads();

    int p = 0;
    for (int k0 = 0; k0 < HIDDEN; k0 += BK) {
        const bool has_next = (k0 + BK) < HIDDEN;
        if (has_next) {            // fill buffer p^1 while computing on p
            loadB(k0 + BK, p ^ 1);
            loadA(k0 + BK, p ^ 1);
        }

        // ---- fragments per documented CDNA5 VGPR layouts, buffer p ----
        v16bf afrag[2], bfrag[2];
        #pragma unroll
        for (int mi = 0; mi < 2; ++mi) {
            // A 16x32 bf16: lane = row (lane&15); K chunks half*8..+7, half*8+16..+23
            const __bf16* ab = &As[p][(waveM * 32 + mi * 16 + l15) * LDA + half * 8];
            bf16x8 c0 = *(const bf16x8*)ab;
            bf16x8 c1 = *(const bf16x8*)(ab + 16);
            #pragma unroll
            for (int e = 0; e < 8; ++e) { afrag[mi][e] = c0[e]; afrag[mi][8 + e] = c1[e]; }
        }
        #pragma unroll
        for (int ni = 0; ni < 2; ++ni) {
            // B 32x16 bf16: lane = column (lane&15); K chunks half*8..+7, half*8+16..+23
            const __bf16* bb = &Bs[p][(waveN * 32 + ni * 16 + l15) * LDB + half * 8];
            bf16x8 c0 = *(const bf16x8*)bb;
            bf16x8 c1 = *(const bf16x8*)(bb + 16);
            #pragma unroll
            for (int e = 0; e < 8; ++e) { bfrag[ni][e] = c0[e]; bfrag[ni][8 + e] = c1[e]; }
        }

        #pragma unroll
        for (int mi = 0; mi < 2; ++mi)
            #pragma unroll
            for (int ni = 0; ni < 2; ++ni)
                acc[mi][ni] = __builtin_amdgcn_wmma_f32_16x16x32_bf16(
                    false, afrag[mi], false, bfrag[ni],
                    (short)0, acc[mi][ni], false, false);

        if (has_next) wait_async0();   // next tile fully resident in LDS
        __syncthreads();
        p ^= 1;
    }

    // ---- epilogue: C/D layout -> m = vgpr + 8*half, n = lane&15 ----
    #pragma unroll
    for (int mi = 0; mi < 2; ++mi)
    #pragma unroll
    for (int ni = 0; ni < 2; ++ni)
    #pragma unroll
    for (int j = 0; j < 8; ++j) {
        const int m = blockM + waveM * 32 + mi * 16 + j + 8 * half;
        const int n = blockN + waveN * 32 + ni * 16 + l15;
        float v = acc[mi][ni][j];
        if (EPI == EPI_XB) {
            // input row m = b*SEQ + t  ->  ws row t*BATCH + b (t-major)
            const int rp = ((m & (SEQ - 1)) << 7) | (m >> 9);
            outBf[(size_t)rp * HIDDEN + n] = (__bf16)v;
        } else if (EPI == EPI_GATE) {
            const float g = 1.0f / (1.0f + __expf(-(v + bg[n])));
            const int rp = ((m & (SEQ - 1)) << 7) | (m >> 9);
            outBf[(size_t)rp * HIDDEN + n] = (__bf16)g;
        } else if (EPI == EPI_STEP) {
            // m is batch index (M == 128 == BATCH)
            const size_t r = (size_t)t * BATCH + m;
            float dtv = __expf(log_dt[n]);
            dtv = fminf(fmaxf(dtv, 0.001f), 0.1f);
            const float xbv = (float)xb[r * HIDDEN + n];
            const float gv  = (float)gate[r * HIDDEN + n];
            float sc = (v + xbv) * dtv;
            sc = (fabsf(sc) > 0.1f) ? sc : 0.0f;
            const float sn = sSrc[(size_t)m * HIDDEN + n] + sc * gv;
            sDst[(size_t)m * HIDDEN + n] = sn;
            outBf[r * HIDDEN + n] = (__bf16)sn;
            if (t == SEQ - 1) stateOut[(size_t)m * HIDDEN + n] = sn;
        } else { // EPI_OUT: Sbuf row m = t*BATCH + b -> ys[b][t][n]
            const int b  = m & (BATCH - 1);
            const int tt = m >> 7;
            outF[(size_t)b * ((size_t)SEQ * HIDDEN) + (size_t)tt * HIDDEN + n] = v;
        }
    }
}

extern "C" void kernel_launch(void* const* d_in, const int* in_sizes, int n_in,
                              void* d_out, int out_size, void* d_ws, size_t ws_size,
                              hipStream_t stream) {
    const float* x      = (const float*)d_in[0];  // (B,S,H)
    const float* state  = (const float*)d_in[1];  // (B,H)
    const float* A      = (const float*)d_in[2];  // (H,H)
    const float* Bm     = (const float*)d_in[3];  // (H,H)
    const float* C      = (const float*)d_in[4];  // (H,H)
    const float* log_dt = (const float*)d_in[5];  // (H,)
    const float* Wg     = (const float*)d_in[6];  // (H,H)
    const float* bg     = (const float*)d_in[7];  // (H,)

    float* ys       = (float*)d_out;                            // B*S*H
    float* stateOut = ys + (size_t)BATCH * SEQ * HIDDEN;        // B*H

    const size_t bigElems = (size_t)BATCH * SEQ * HIDDEN;       // 67,108,864
    const size_t wElems   = (size_t)HIDDEN * HIDDEN;            // 1,048,576

    char* ws = (char*)d_ws;
    __bf16* xbBuf  = (__bf16*)ws; ws += bigElems * sizeof(__bf16);  // 128 MB
    __bf16* gBuf   = (__bf16*)ws; ws += bigElems * sizeof(__bf16);  // 128 MB
    __bf16* sSnap  = (__bf16*)ws; ws += bigElems * sizeof(__bf16);  // 128 MB
    __bf16* Abf    = (__bf16*)ws; ws += wElems * sizeof(__bf16);
    __bf16* Bmbf   = (__bf16*)ws; ws += wElems * sizeof(__bf16);
    __bf16* Cbf    = (__bf16*)ws; ws += wElems * sizeof(__bf16);
    __bf16* Wgbf   = (__bf16*)ws; ws += wElems * sizeof(__bf16);
    float*  s0     = (float*)ws;  ws += (size_t)BATCH * HIDDEN * sizeof(float);
    float*  s1     = (float*)ws;  ws += (size_t)BATCH * HIDDEN * sizeof(float);

    // --- convert weights to bf16 once ---
    {
        const int threads = 256;
        const int blocks  = (int)((wElems + threads - 1) / threads);
        cvt_f32_to_bf16<<<blocks, threads, 0, stream>>>(A,  Abf,  (int)wElems);
        cvt_f32_to_bf16<<<blocks, threads, 0, stream>>>(Bm, Bmbf, (int)wElems);
        cvt_f32_to_bf16<<<blocks, threads, 0, stream>>>(C,  Cbf,  (int)wElems);
        cvt_f32_to_bf16<<<blocks, threads, 0, stream>>>(Wg, Wgbf, (int)wElems);
    }

    // --- Pass 1: XB and gate over all (b,t) rows (parallel, saturating) ---
    dim3 gridBig(HIDDEN / BN, (BATCH * SEQ) / BM);   // 16 x 512
    ssm_gemm<EPI_XB, false><<<gridBig, 256, 0, stream>>>(
        x, nullptr, Bmbf, nullptr, nullptr, nullptr, nullptr,
        nullptr, nullptr, nullptr, xbBuf, nullptr, 0);
    ssm_gemm<EPI_GATE, false><<<gridBig, 256, 0, stream>>>(
        x, nullptr, Wgbf, nullptr, nullptr, nullptr, nullptr,
        bg, nullptr, nullptr, gBuf, nullptr, 0);

    // --- init ping-pong state ---
    hipMemcpyAsync(s0, state, (size_t)BATCH * HIDDEN * sizeof(float),
                   hipMemcpyDeviceToDevice, stream);

    // --- Pass 2: serial scan, one latency-bound launch per timestep ---
    dim3 gridStep(HIDDEN / BN, 1);                   // 16 x 1 (M = 128 = BM)
    for (int t = 0; t < SEQ; ++t) {
        const float* ss = (t & 1) ? s1 : s0;
        float*       sd = (t & 1) ? s0 : s1;
        ssm_gemm<EPI_STEP, false><<<gridStep, 256, 0, stream>>>(
            ss, nullptr, Abf, xbBuf, gBuf, ss, sd,
            nullptr, log_dt, nullptr, sSnap, stateOut, t);
    }

    // --- Pass 3: ys = Sbuf @ C^T (parallel, saturating) ---
    ssm_gemm<EPI_OUT, true><<<gridBig, 256, 0, stream>>>(
        nullptr, sSnap, Cbf, nullptr, nullptr, nullptr, nullptr,
        nullptr, nullptr, ys, nullptr, nullptr, 0);
}
